// DR2FWL2ConvSimple_81372450390513
// MI455X (gfx1250) — matrix operand order repaired and checked by reading, not compiled, not run on hardware
//
#include <hip/hip_runtime.h>
#include <hip/hip_bf16.h>

#define CH 128
#define N0 20000
#define E1 500000
#define E2 500000
#define TT 500000

typedef __attribute__((ext_vector_type(16))) __bf16 v16bf;
typedef __attribute__((ext_vector_type(8)))  float  v8f;

// ---------------------------------------------------------------------------
// Fused: row build (identity / gather / gather2-sum) -> Linear(x @ W^T + b)
// via bf16 WMMA -> epilogue (optional ReLU) -> store / add / atomic scatter.
// Block = 256 threads = 8 waves; each wave computes 16 rows x 128 outs.
// LDS holds W and row tiles in *fragment-major* swizzled layout so every
// lane's 16-bf16 fragment is 32 contiguous bytes (2x ds_load_b128).
// OUTMODE: 0 = store, 1 = add-in-place, 2 = atomicAdd(Out[sidx[r]])
// ---------------------------------------------------------------------------
template<int RELU, int OUTMODE>
__global__ __launch_bounds__(256)
void wmma_lin_kernel(const float* __restrict__ X, const int* __restrict__ idxA,
                     const float* __restrict__ Y, const int* __restrict__ idxB,
                     const float* __restrict__ W, const float* __restrict__ bias,
                     float* __restrict__ Out, const int* __restrict__ sidx,
                     int nRows)
{
    // B-fragment-major W: [nt(8)][ks(4)][lane(32)][elem(16)]  (32 KB)
    __shared__ __attribute__((aligned(32))) __bf16 sW[CH * CH];
    // A-fragment-major rows: [wave(8)][ks(4)][lane(32)][elem(16)]  (32 KB)
    __shared__ __attribute__((aligned(32))) __bf16 sX[128 * CH];
    __shared__ float sBias[CH];

    const int tid  = threadIdx.x;
    const int lane = tid & 31;
    const int wave = tid >> 5;
    const int rowBlock = blockIdx.x * 128;

    // ---- Stage W^T into LDS in B-fragment layout (dest-major, float4 src) ----
    // B layout: lane = ((kk>>3)&1)*16 + (n&15), elem = (kk&7) + ((kk>>4)&1)*8, tile nt = n>>4, ks = k>>5
    for (int d = tid * 4; d < CH * CH; d += 256 * 4) {
        int idx   = d & 15;
        int lanef = (d >> 4) & 31;
        int ks    = (d >> 9) & 3;
        int nt    = d >> 11;
        int k     = ks * 32 + (lanef >> 4) * 8 + (idx & 7) + (idx >> 3) * 16;   // idx in {0,4,8,12}
        int n     = nt * 16 + (lanef & 15);
        float4 w4 = *(const float4*)&W[n * CH + k];
        __bf16* dst = &sW[d];
        dst[0] = (__bf16)w4.x; dst[1] = (__bf16)w4.y;
        dst[2] = (__bf16)w4.z; dst[3] = (__bf16)w4.w;
    }
    if (tid < CH) sBias[tid] = bias[tid];

    // ---- Stage this wave's 16 input rows directly into A-fragment layout ----
    // A layout: lane = ((cc>>3)&1)*16 + m, elem = (cc&7) + ((cc>>4)&1)*8
    const int baseRow = rowBlock + wave * 16;
    {
        const int c0    = lane * 4;          // this lane covers cols c0..c0+3
        const int ks    = c0 >> 5;
        const int cc    = c0 & 31;
        const int hgrp  = (cc >> 3) & 1;
        const int idx0  = (cc & 7) + ((cc >> 4) & 1) * 8;
        for (int r = 0; r < 16; ++r) {
            int gr = baseRow + r;
            float4 v = make_float4(0.f, 0.f, 0.f, 0.f);
            if (gr < nRows) {
                long ra = idxA ? (long)idxA[gr] : (long)gr;
                v = ((const float4*)(X + ra * CH))[lane];
                if (Y) {
                    long rb = idxB ? (long)idxB[gr] : (long)gr;
                    float4 w4 = ((const float4*)(Y + rb * CH))[lane];
                    v.x += w4.x; v.y += w4.y; v.z += w4.z; v.w += w4.w;
                }
            }
            int lanef = hgrp * 16 + r;
            __bf16* dst = &sX[((wave * 4 + ks) * 32 + lanef) * 16 + idx0];
            dst[0] = (__bf16)v.x; dst[1] = (__bf16)v.y;
            dst[2] = (__bf16)v.z; dst[3] = (__bf16)v.w;
        }
    }
    __syncthreads();

    // ---- Load 4 A fragments: each is 32 contiguous bytes per lane ----
    v16bf afrag[4];
    #pragma unroll
    for (int ks = 0; ks < 4; ++ks)
        afrag[ks] = *(const v16bf*)&sX[((wave * 4 + ks) * 32 + lane) * 16];

    // Wave-uniform: is this a full 16-row tile? (true for all but the last block)
    const bool fullTile = (baseRow + 16) <= nRows;
    const int  nlanebase = lane & 15;
    const int  mbase     = 8 * (lane >> 4);

    // ---- 8 N-tiles x 4 K-steps of v_wmma_f32_16x16x32_bf16 ----
    #pragma unroll
    for (int nt = 0; nt < 8; ++nt) {
        const int nlane = nt * 16 + nlanebase;
        v8f acc = {};
        #pragma unroll
        for (int ks = 0; ks < 4; ++ks) {
            v16bf bfrag = *(const v16bf*)&sW[((nt * 4 + ks) * 32 + lane) * 16];
            acc = __builtin_amdgcn_wmma_f32_16x16x32_bf16(
                      false, afrag[ks], false, bfrag,
                      (short)0, acc, false, false);
        }
        // ---- Epilogue: bias, ReLU ----
        const float b = sBias[nlane];
        float vals[8];
        #pragma unroll
        for (int v = 0; v < 8; ++v) {
            float t = acc[v] + b;
            if (RELU) t = fmaxf(t, 0.f);
            vals[v] = t;
        }
        // ---- Writeout ----
        if (OUTMODE == 2) {
            if (fullTile) {
                #pragma unroll
                for (int v = 0; v < 8; ++v) {
                    int gr = baseRow + mbase + v;
                    atomicAdd(&Out[(long)sidx[gr] * CH + nlane], vals[v]);
                }
            } else {
                for (int v = 0; v < 8; ++v) {
                    int gr = baseRow + mbase + v;
                    if (gr < nRows)
                        atomicAdd(&Out[(long)sidx[gr] * CH + nlane], vals[v]);
                }
            }
        } else {
            float* outp = Out + (long)(baseRow + mbase) * CH + nlane;
            if (fullTile) {
                #pragma unroll
                for (int v = 0; v < 8; ++v) {
                    if (OUTMODE == 0) outp[(long)v * CH] = vals[v];
                    else              outp[(long)v * CH] += vals[v];
                }
            } else {
                for (int v = 0; v < 8; ++v) {
                    if (baseRow + mbase + v < nRows) {
                        if (OUTMODE == 0) outp[(long)v * CH] = vals[v];
                        else              outp[(long)v * CH] += vals[v];
                    }
                }
            }
        }
    }
}

// ---------------------------------------------------------------------------
// Elementwise helpers
// ---------------------------------------------------------------------------
__global__ void fill0_kernel(float* p, long n) {
    long i = (long)blockIdx.x * blockDim.x + threadIdx.x;
    long stride = (long)gridDim.x * blockDim.x;
    for (; i < n; i += stride) p[i] = 0.f;
}
__global__ void copy_kernel(float* __restrict__ dst, const float* __restrict__ src, long n) {
    long i = (long)blockIdx.x * blockDim.x + threadIdx.x;
    long stride = (long)gridDim.x * blockDim.x;
    for (; i < n; i += stride) dst[i] = src[i];
}
__global__ void axpy_kernel(float* __restrict__ dst, const float* __restrict__ src, long n) {
    long i = (long)blockIdx.x * blockDim.x + threadIdx.x;
    long stride = (long)gridDim.x * blockDim.x;
    for (; i < n; i += stride) dst[i] += src[i];
}

// ---------------------------------------------------------------------------
// Host-side dispatch
// ---------------------------------------------------------------------------
static void lin(const float* X, const int* iA, const float* Y, const int* iB,
                const float* W, const float* b, float* Out, const int* sidx,
                int nRows, int relu, int outmode, hipStream_t st)
{
    dim3 grid((unsigned)((nRows + 127) / 128)), blk(256);
    if (relu) {
        if (outmode == 0)      wmma_lin_kernel<1,0><<<grid, blk, 0, st>>>(X,iA,Y,iB,W,b,Out,sidx,nRows);
        else if (outmode == 1) wmma_lin_kernel<1,1><<<grid, blk, 0, st>>>(X,iA,Y,iB,W,b,Out,sidx,nRows);
        else                   wmma_lin_kernel<1,2><<<grid, blk, 0, st>>>(X,iA,Y,iB,W,b,Out,sidx,nRows);
    } else {
        if (outmode == 0)      wmma_lin_kernel<0,0><<<grid, blk, 0, st>>>(X,iA,Y,iB,W,b,Out,sidx,nRows);
        else if (outmode == 1) wmma_lin_kernel<0,1><<<grid, blk, 0, st>>>(X,iA,Y,iB,W,b,Out,sidx,nRows);
        else                   wmma_lin_kernel<0,2><<<grid, blk, 0, st>>>(X,iA,Y,iB,W,b,Out,sidx,nRows);
    }
}

static void fill0(float* p, long n, hipStream_t st) {
    fill0_kernel<<<2048, 256, 0, st>>>(p, n);
}

extern "C" void kernel_launch(void* const* d_in, const int* in_sizes, int n_in,
                              void* d_out, int out_size, void* d_ws, size_t ws_size,
                              hipStream_t stream)
{
    (void)in_sizes; (void)n_in; (void)out_size; (void)ws_size;
    const float* x0     = (const float*)d_in[0];
    const float* x1     = (const float*)d_in[1];
    const float* x2     = (const float*)d_in[2];
    const int*   ei1    = (const int*)d_in[3];   // [2, E1]
    const int*   ei2    = (const int*)d_in[4];   // [2, E2]
    const int*   tri111 = (const int*)d_in[5];   // [3, T]
    const int*   tri222 = (const int*)d_in[6];
    const int*   tri112 = (const int*)d_in[7];
    const int*   inv1   = (const int*)d_in[8];
    /* inv2 = d_in[9] unused by reference */
    const float* innW = (const float*)d_in[10]; const float* innB = (const float*)d_in[11];
    const float* W111 = (const float*)d_in[12]; const float* b111 = (const float*)d_in[13];
    const float* W222 = (const float*)d_in[14]; const float* b222 = (const float*)d_in[15];
    const float* W211 = (const float*)d_in[16]; const float* b211 = (const float*)d_in[17];
    const float* m0aW = (const float*)d_in[18]; const float* m0aB = (const float*)d_in[19];
    const float* m0bW = (const float*)d_in[20]; const float* m0bB = (const float*)d_in[21];
    const float* m1aW = (const float*)d_in[22]; const float* m1aB = (const float*)d_in[23];
    const float* m1bW = (const float*)d_in[24]; const float* m1bB = (const float*)d_in[25];
    const float* m2aW = (const float*)d_in[26]; const float* m2aB = (const float*)d_in[27];
    const float* m2bW = (const float*)d_in[28]; const float* m2bB = (const float*)d_in[29];

    float* out = (float*)d_out;
    float* o0 = out;
    float* o1 = o0 + (long)N0 * CH;
    float* o2 = o1 + (long)E1 * CH;

    float* a0  = (float*)d_ws;
    float* a1  = a0 + (long)N0 * CH;
    float* a2  = a1 + (long)E1 * CH;
    float* tmp = a2 + (long)E2 * CH;   // reused sequentially, max(E1,E2) rows

    const int* s1 = ei1, *e1 = ei1 + E1;
    const int* s2 = ei2, *e2 = ei2 + E2;
    const int* t111_0 = tri111, *t111_1 = tri111 + TT, *t111_2 = tri111 + 2L*TT;
    const int* t222_0 = tri222, *t222_1 = tri222 + TT, *t222_2 = tri222 + 2L*TT;
    const int* t112_0 = tri112, *t112_1 = tri112 + TT, *t112_2 = tri112 + 2L*TT;

    // ---- zero accumulators ----
    fill0(a0, (long)N0 * CH, stream);
    fill0(a1, (long)E1 * CH, stream);
    fill0(a2, (long)E2 * CH, stream);

    // ---- aggr (0,1,1) ----
    lin(x0, s1, x0, e1, innW, innB, a1, nullptr, E1, 1, 0, stream);     // a1 = inner(x0[s]+x0[e])
    lin(x1, nullptr, nullptr, nullptr, innW, innB, a0, s1, E1, 1, 2, stream); // a0 += seg(inner(x1), s)

    // ---- aggr (0,2,2) ----
    lin(x0, s2, x0, e2, innW, innB, a2, nullptr, E2, 1, 0, stream);
    lin(x2, nullptr, nullptr, nullptr, innW, innB, a0, s2, E2, 1, 2, stream);

    // ---- aggr (1,1,1) ----
    fill0(tmp, (long)E1 * CH, stream);
    lin(x1, t111_1, x1, t111_2, innW, innB, tmp, t111_0, TT, 1, 2, stream);
    lin(tmp, nullptr, nullptr, nullptr, W111, b111, a1, nullptr, E1, 0, 1, stream);

    // ---- aggr (2,2,2) ----
    fill0(tmp, (long)E2 * CH, stream);
    lin(x2, t222_1, x2, t222_2, innW, innB, tmp, t222_0, TT, 1, 2, stream);
    lin(tmp, nullptr, nullptr, nullptr, W222, b222, a2, nullptr, E2, 0, 1, stream);

    // ---- aggr (1,1,2): a2 term ----
    fill0(tmp, (long)E2 * CH, stream);
    lin(x1, t112_0, x1, t112_1, innW, innB, tmp, t112_2, TT, 1, 2, stream);
    lin(tmp, nullptr, nullptr, nullptr, W211, b211, a2, nullptr, E2, 0, 1, stream);

    // ---- aggr (1,1,2): a1 term (out_iji) ----
    fill0(tmp, (long)E1 * CH, stream);
    lin(x1, t112_1, x2, t112_2, innW, innB, tmp, t112_0, TT, 1, 2, stream);
    // a1 += lin(out_iji + out_iji[inv1], l211)
    lin(tmp, nullptr, tmp, inv1, W211, b211, a1, nullptr, E1, 0, 1, stream);

    // ---- z = x + a (eps = 0) ----
    axpy_kernel<<<2048, 256, 0, stream>>>(a0, x0, (long)N0 * CH);
    axpy_kernel<<<2048, 256, 0, stream>>>(a1, x1, (long)E1 * CH);
    axpy_kernel<<<2048, 256, 0, stream>>>(a2, x2, (long)E2 * CH);

    // ---- o0 = mlp(z0) + x0 ----
    copy_kernel<<<2048, 256, 0, stream>>>(o0, x0, (long)N0 * CH);
    lin(a0, nullptr, nullptr, nullptr, m0aW, m0aB, tmp, nullptr, N0, 1, 0, stream);
    lin(tmp, nullptr, nullptr, nullptr, m0bW, m0bB, o0, nullptr, N0, 0, 1, stream);

    // ---- o1 = mlp(z1) + x1 ----
    copy_kernel<<<2048, 256, 0, stream>>>(o1, x1, (long)E1 * CH);
    lin(a1, nullptr, nullptr, nullptr, m1aW, m1aB, tmp, nullptr, E1, 1, 0, stream);
    lin(tmp, nullptr, nullptr, nullptr, m1bW, m1bB, o1, nullptr, E1, 0, 1, stream);

    // ---- o2 = mlp(z2) + x2 ----
    copy_kernel<<<2048, 256, 0, stream>>>(o2, x2, (long)E2 * CH);
    lin(a2, nullptr, nullptr, nullptr, m2aW, m2aB, tmp, nullptr, E2, 1, 0, stream);
    lin(tmp, nullptr, nullptr, nullptr, m2bW, m2bB, o2, nullptr, E2, 0, 1, stream);
}
